// TileTask_62277025792545
// MI455X (gfx1250) — 8-bit matrix operand order repaired and checked by reading, compiled, not run on hardware
//
#include <hip/hip_runtime.h>
#include <stdint.h>
#include <stddef.h>

typedef int v8i __attribute__((ext_vector_type(8)));

#define BB   4
#define CIN  64
#define COUT 64
#define HH   128
#define WW   128
#define HP   130
#define WP   130

__global__ void quant_pad_kernel(const float* __restrict__ x, uint8_t* __restrict__ xq) {
  int idx = blockIdx.x * blockDim.x + threadIdx.x;
  const int total = BB * HP * WP * CIN;
  if (idx >= total) return;
  int c  = idx & 63;
  int t  = idx >> 6;
  int wp = t % WP; t /= WP;
  int hp = t % HP;
  int b  = t / HP;
  uint8_t v = 0;
  if (hp >= 1 && hp <= HH && wp >= 1 && wp <= WW) {
    float f = x[(((size_t)b * CIN + c) * HH + (hp - 1)) * WW + (wp - 1)];
    int q = (int)__builtin_rintf(f);
    q = q < -128 ? -128 : (q > 127 ? 127 : q);
    v = (uint8_t)(int8_t)q;
  }
  xq[idx] = v;
}

__global__ void pack_weights_kernel(const float* __restrict__ w, int* __restrict__ bpack) {
  int idx = blockIdx.x * blockDim.x + threadIdx.x;
  if (idx >= 9 * 4 * 8 * 32) return;
  int lane  = idx & 31;
  int v     = (idx >> 5) & 7;
  int ntile = (idx >> 8) & 3;
  int k9    = idx >> 10;
  int ky = k9 / 3, kx = k9 % 3;
  int n     = ntile * 16 + (lane & 15);
  int kbase = ((v >> 1) * 16) + ((v & 1) * 4) + ((lane >> 4) * 8);
  unsigned pack = 0;
  for (int j = 0; j < 4; ++j) {
    int c = kbase + j;
    float f = w[(((size_t)n * CIN + c) * 3 + ky) * 3 + kx];
    int q = (int)__builtin_rintf(f);
    pack |= ((unsigned)(uint8_t)(int8_t)q) << (8 * j);
  }
  bpack[idx] = (int)pack;
}

__global__ __launch_bounds__(128)
void conv_wmma_kernel(const uint8_t* __restrict__ xq, const int* __restrict__ bpack,
                      const float* __restrict__ bias, float* __restrict__ convout) {
  const int lane   = threadIdx.x & 31;
  const int ntile  = threadIdx.x >> 5;
  const int mtile  = blockIdx.x;
  const int wt = mtile & 7;
  const int h  = (mtile >> 3) & (HH - 1);
  const int b  = mtile >> 10;
  const int w0 = wt * 16;
  const int laneLo = lane & 15;
  const int laneHi = lane >> 4;

  v8i acc[8] = {};

  const int cpair0 = laneHi * 8;

  for (int ky = 0; ky < 3; ++ky) {
    const uint8_t* rowp = xq + ((size_t)(b * HP + (h + ky)) * WP) * CIN;
    for (int kx = 0; kx < 3; ++kx) {
      const int k9 = ky * 3 + kx;

      const int* bp = bpack + ((k9 * 4 + ntile) * 8) * 32 + lane;
      v8i bfrag;
      #pragma unroll
      for (int v = 0; v < 8; ++v) bfrag[v] = bp[v * 32];

      const uint8_t* pix = rowp + (size_t)(w0 + laneLo + kx) * CIN + cpair0;
      unsigned xfrag[8];
      #pragma unroll
      for (int q = 0; q < 4; ++q) {
        uint64_t d = *(const uint64_t*)(pix + q * 16);
        xfrag[2 * q]     = (unsigned)d;
        xfrag[2 * q + 1] = (unsigned)(d >> 32);
      }

      #pragma unroll
      for (int p = 0; p < 8; ++p) {
        v8i afrag;
        #pragma unroll
        for (int v = 0; v < 8; ++v)
          afrag[v] = (int)((xfrag[v] >> p) & 0x01010101u);
        acc[p] = __builtin_amdgcn_wmma_i32_16x16x64_iu8(
             false, afrag,  true, bfrag, acc[p],
             false,  false);
      }
    }
  }

  const int   n  = ntile * 16 + laneLo;
  const float bn = bias[n];
  const float planew[8] = {1.f, 2.f, 4.f, 8.f, 16.f, 32.f, 64.f, -128.f};
  #pragma unroll
  for (int r = 0; r < 8; ++r) {
    const int m = r + laneHi * 8;
    float y = 0.f;
    #pragma unroll
    for (int p = 0; p < 8; ++p) {
      float t = __builtin_rintf((float)acc[p][r] * 0.5f);
      t = fminf(fmaxf(t, -128.f), 127.f);
      y += planew[p] * (2.f * t);
    }
    y += bn;
    convout[(((size_t)(b * COUT + n)) * HH + h) * WW + (w0 + m)] = y;
  }
}

__global__ void post_pool_kernel(const float* __restrict__ conv,
                                 const float* __restrict__ merge,
                                 const float* __restrict__ gather,
                                 const float* __restrict__ ctrans_i,
                                 const float* __restrict__ ctrans_s,
                                 const float* __restrict__ strans_i,
                                 const float* __restrict__ strans_s,
                                 const float* __restrict__ atrans_i,
                                 const float* __restrict__ atrans_s,
                                 float* __restrict__ out) {
  const int HO = HH / 2, WO = WW / 2;
  int idx = blockIdx.x * blockDim.x + threadIdx.x;
  if (idx >= BB * COUT * HO * WO) return;
  int wo = idx % WO; int t = idx / WO;
  int ho = t % HO;  t /= HO;
  int n  = t % COUT;
  int b  = t / COUT;

  const float cmult = ctrans_i[n] * exp2f(ctrans_s[n]);
  const float smult = strans_i[0] * exp2f(strans_s[0]);
  const float amult = atrans_i[0] * exp2f(atrans_s[0]);

  float best = -3.4e38f;
  #pragma unroll
  for (int dy = 0; dy < 2; ++dy)
    #pragma unroll
    for (int dx = 0; dx < 2; ++dx) {
      size_t off = (((size_t)(b * COUT + n)) * HH + (2 * ho + dy)) * WW + (2 * wo + dx);
      float x = conv[off] + merge[off];
      x = __builtin_rintf(x * cmult);
      x = fminf(fmaxf(x, -128.f), 127.f);
      x = x + gather[off];
      x = __builtin_rintf(x * smult);
      x = fminf(fmaxf(x, -128.f), 127.f);
      x = fmaxf(x, 0.f);
      x = __builtin_rintf(x * amult);
      x = fminf(fmaxf(x, -128.f), 127.f);
      best = fmaxf(best, x);
    }
  out[idx] = best;
}

extern "C" void kernel_launch(void* const* d_in, const int* in_sizes, int n_in,
                              void* d_out, int out_size, void* d_ws, size_t ws_size,
                              hipStream_t stream) {
  const float* cast_in   = (const float*)d_in[0];
  const float* merge_in  = (const float*)d_in[1];
  const float* gather_in = (const float*)d_in[2];
  const float* conv_w    = (const float*)d_in[3];
  const float* conv_b    = (const float*)d_in[4];
  const float* ctrans_i  = (const float*)d_in[5];
  const float* ctrans_s  = (const float*)d_in[6];
  const float* strans_i  = (const float*)d_in[7];
  const float* strans_s  = (const float*)d_in[8];
  const float* atrans_i  = (const float*)d_in[9];
  const float* atrans_s  = (const float*)d_in[10];
  (void)in_sizes; (void)n_in; (void)out_size; (void)ws_size;

  uint8_t* ws = (uint8_t*)d_ws;
  uint8_t* xq = ws;
  size_t off = ((size_t)BB * HP * WP * CIN + 255) & ~(size_t)255;
  int* bpack = (int*)(ws + off);
  off += (size_t)9 * 4 * 8 * 32 * sizeof(int);
  off  = (off + 255) & ~(size_t)255;
  float* convout = (float*)(ws + off);

  {
    int total = BB * HP * WP * CIN;
    quant_pad_kernel<<<(total + 255) / 256, 256, 0, stream>>>(cast_in, xq);
  }
  {
    int total = 9 * 4 * 8 * 32;
    pack_weights_kernel<<<(total + 255) / 256, 256, 0, stream>>>(conv_w, bpack);
  }
  conv_wmma_kernel<<<4096, 128, 0, stream>>>(xq, bpack, conv_b, convout);
  {
    int total = BB * COUT * (HH / 2) * (WW / 2);
    post_pool_kernel<<<(total + 255) / 256, 256, 0, stream>>>(
        convout, merge_in, gather_in, ctrans_i, ctrans_s,
        strans_i, strans_s, atrans_i, atrans_s, (float*)d_out);
  }
}
